// RNN_8856222564555
// MI455X (gfx1250) — compile-verified, matrix operand order checked by reading, not run on hardware
//
#include <hip/hip_runtime.h>
#include <hip/hip_bf16.h>

#define BATCH  64
#define SEQ    512
#define EMBED  256
#define HIDDEN 512
#define BS     (BATCH * SEQ)

typedef __attribute__((ext_vector_type(16))) __bf16 v16bf;
typedef __attribute__((ext_vector_type(8)))  float  v8f;

union FragU { uint4 q[2]; v16bf v; };

__device__ __forceinline__ unsigned short f32_to_bf16(float f) {
  unsigned u = __float_as_uint(f);
  u += 0x7FFFu + ((u >> 16) & 1u);   // round-to-nearest-even
  return (unsigned short)(u >> 16);
}

// 16x32 bf16 fragment: lane L<16 -> row M=L, K-chunks {k0..k0+7},{k0+16..k0+23};
// lane L>=16 -> row M=L-16, K-chunks shifted by 8.  Two b128 loads per lane.
__device__ __forceinline__ v16bf load_frag(const unsigned short* base, int ld,
                                           int row0, int k0, int lane) {
  const int r  = row0 + (lane & 15);
  const int kk = k0 + ((lane >> 4) << 3);
  const unsigned short* p = base + (size_t)r * ld + kk;
  FragU f;
  f.q[0] = *(const uint4*)(p);
  f.q[1] = *(const uint4*)(p + 16);
  return f.v;
}

__device__ __forceinline__ v8f wmma_bf16(v16bf a, v16bf b, v8f c) {
  return __builtin_amdgcn_wmma_f32_16x16x32_bf16(false, a, false, b,
                                                 (short)0, c, false, false);
}

// ---------------------------------------------------------------------------
// C = A[M,K]bf16 * W[N,K]bf16^T + bias[N], M=BS, N=HIDDEN.
// Wave tile = 32x64 (2 A-frags x 4 B-frags, 8 wmma per K-step; B reused 2x).
// Block = 256 threads = 8 waves stacked in M.  Grid = (BS/256, HIDDEN/64).
// out_sbh: 0 -> C[m][n] row-major; 1 -> C[s][BATCH][n] (time-major for scan).
// ---------------------------------------------------------------------------
__global__ __launch_bounds__(256) void gemm_bf16_bias(
    const unsigned short* __restrict__ A, const unsigned short* __restrict__ W,
    const float* __restrict__ bias, float* __restrict__ C, int K, int out_sbh) {
  const int lane = threadIdx.x & 31;
  const int wave = threadIdx.x >> 5;
  const int m0 = (blockIdx.x * 8 + wave) * 32;
  const int n0 = blockIdx.y * 64;
  v8f acc[2][4] = {};
  for (int k0 = 0; k0 < K; k0 += 32) {
    v16bf a0 = load_frag(A, K, m0,      k0, lane);
    v16bf a1 = load_frag(A, K, m0 + 16, k0, lane);
#pragma unroll
    for (int j = 0; j < 4; ++j) {
      v16bf b = load_frag(W, K, n0 + 16 * j, k0, lane);
      acc[0][j] = wmma_bf16(a0, b, acc[0][j]);
      acc[1][j] = wmma_bf16(a1, b, acc[1][j]);
    }
  }
  const int col = lane & 15;
  const int rb  = (lane >> 4) << 3;
#pragma unroll
  for (int i = 0; i < 2; ++i) {
#pragma unroll
    for (int j = 0; j < 4; ++j) {
      const int n  = n0 + 16 * j + col;
      const float bv = bias[n];
#pragma unroll
      for (int r = 0; r < 8; ++r) {
        const int m = m0 + 16 * i + rb + r;
        size_t idx;
        if (out_sbh) {              // m = b*SEQ + s  ->  [s][b][n]
          const int b = m >> 9;     // SEQ = 512
          const int s = m & 511;
          idx = ((size_t)s * BATCH + b) * HIDDEN + n;
        } else {
          idx = (size_t)m * HIDDEN + n;
        }
        C[idx] = acc[i][j][r] + bv;
      }
    }
  }
}

// ---------------------------------------------------------------------------
// Recurrence: h_t = tanh(xp_t + h_{t-1} @ Whh^T) per 16-batch tile.
// Block = 512 threads = 16 waves; wave w owns hidden cols [32w, 32w+32).
// LDS: W_hh K-low half (256KB, preloaded) + double-buffered h (2x16KB).
// xp is time-major [S][B][H]: per-step slab is one contiguous 32KB block,
// prefetched one step ahead.  One barrier per step (double buffer).
// ---------------------------------------------------------------------------
template <bool STORE_Y, bool ACC_POOL>
__global__ __launch_bounds__(512) void rnn_recurrence(
    const float* __restrict__ xp, const unsigned short* __restrict__ Whh,
    unsigned short* __restrict__ y, float* __restrict__ pooled) {
  extern __shared__ unsigned short smem[];
  unsigned short* wlds = smem;                       // [HIDDEN][256] K-low half
  unsigned short* hbuf = smem + HIDDEN * 256;        // [2][16][HIDDEN]

  const int b0   = blockIdx.x * 16;
  const int lane = threadIdx.x & 31;
  const int wave = threadIdx.x >> 5;                 // 0..15
  const int n0   = wave * 32;

  // Preload W_hh[:, 0:256) into LDS (row-major, ld=256).
  {
    const uint4* src = (const uint4*)Whh;            // 64 uint4 per 512-half row
    uint4*       dst = (uint4*)wlds;                 // 32 uint4 per 256-half row
    for (int i = threadIdx.x; i < HIDDEN * 32; i += blockDim.x)
      dst[i] = src[(i >> 5) * 64 + (i & 31)];
  }
  for (int i = threadIdx.x; i < 2 * 16 * HIDDEN; i += blockDim.x) hbuf[i] = 0;
  __syncthreads();

  const int col   = lane & 15;
  const int rb    = (lane >> 4) << 3;
  const int ar    = lane & 15;
  const int akoff = (lane >> 4) << 3;
  float psum[2][8] = {};
  int cur = 0;

  for (int t = 0; t < SEQ; ++t) {
    unsigned short* hc = hbuf + cur * (16 * HIDDEN);
    unsigned short* hn = hbuf + (cur ^ 1) * (16 * HIDDEN);
    v8f acc[2] = {};

    // K in [0,256): B fragments from LDS.
    for (int k0 = 0; k0 < 256; k0 += 32) {
      const int ak = k0 + akoff;
      FragU a;
      a.q[0] = *(const uint4*)&hc[ar * HIDDEN + ak];
      a.q[1] = *(const uint4*)&hc[ar * HIDDEN + ak + 16];
      FragU bA, bB;
      const int r0 = n0 + (lane & 15);
      bA.q[0] = *(const uint4*)&wlds[r0 * 256 + ak];
      bA.q[1] = *(const uint4*)&wlds[r0 * 256 + ak + 16];
      bB.q[0] = *(const uint4*)&wlds[(r0 + 16) * 256 + ak];
      bB.q[1] = *(const uint4*)&wlds[(r0 + 16) * 256 + ak + 16];
      acc[0] = wmma_bf16(a.v, bA.v, acc[0]);
      acc[1] = wmma_bf16(a.v, bB.v, acc[1]);
    }
    // K in [256,512): B fragments from global (L2-resident).
    for (int k0 = 256; k0 < 512; k0 += 32) {
      const int ak = k0 + akoff;
      FragU a;
      a.q[0] = *(const uint4*)&hc[ar * HIDDEN + ak];
      a.q[1] = *(const uint4*)&hc[ar * HIDDEN + ak + 16];
      v16bf bA = load_frag(Whh, HIDDEN, n0,      k0, lane);
      v16bf bB = load_frag(Whh, HIDDEN, n0 + 16, k0, lane);
      acc[0] = wmma_bf16(a.v, bA, acc[0]);
      acc[1] = wmma_bf16(a.v, bB, acc[1]);
    }

    // Prefetch next step's 32KB xp slab while this step finishes.
    if (t + 1 < SEQ) {
      const char* pf =
          (const char*)(xp + ((size_t)(t + 1) * BATCH + b0) * HIDDEN);
      __builtin_prefetch(pf + (int)threadIdx.x * 64, 0, 0);
    }

    const float* xpt = xp + ((size_t)t * BATCH + b0) * HIDDEN;
#pragma unroll
    for (int j = 0; j < 2; ++j) {
      const int n = n0 + 16 * j + col;
#pragma unroll
      for (int r = 0; r < 8; ++r) {
        const int m = rb + r;
        const float xv = xpt[(size_t)m * HIDDEN + n];
        const float hv = tanhf(acc[j][r] + xv);
        const unsigned short hb = f32_to_bf16(hv);
        hn[m * HIDDEN + n] = hb;
        if (STORE_Y)  y[((size_t)(b0 + m) * SEQ + t) * HIDDEN + n] = hb;
        if (ACC_POOL) psum[j][r] += hv;
      }
    }
    __syncthreads();   // h_t visible; previous buffer free for reuse
    cur ^= 1;
  }

  if (ACC_POOL) {
#pragma unroll
    for (int j = 0; j < 2; ++j) {
      const int n = n0 + 16 * j + col;
#pragma unroll
      for (int r = 0; r < 8; ++r)
        pooled[(size_t)(b0 + rb + r) * HIDDEN + n] = psum[j][r] * (1.0f / SEQ);
    }
  }
}

// ---------------------------------------------------------------------------
// Prep + head kernels
// ---------------------------------------------------------------------------
__global__ void embed_gather_bf16(const int* __restrict__ x,
                                  const float* __restrict__ emb,
                                  unsigned short* __restrict__ out) {
  const int i   = blockIdx.x * blockDim.x + threadIdx.x;  // BS*EMBED/4 threads
  const int row = i >> 6;                                 // 64 float4 per row
  const int c4  = (i & 63) << 2;
  const int tok = x[row];
  const float4 v = *(const float4*)(emb + (size_t)tok * EMBED + c4);
  uint2 pk;
  pk.x = (unsigned)f32_to_bf16(v.x) | ((unsigned)f32_to_bf16(v.y) << 16);
  pk.y = (unsigned)f32_to_bf16(v.z) | ((unsigned)f32_to_bf16(v.w) << 16);
  *(uint2*)(out + (size_t)row * EMBED + c4) = pk;
}

__global__ void cvt_f32_bf16(const float* __restrict__ in,
                             unsigned short* __restrict__ out, int n) {
  const int i = blockIdx.x * blockDim.x + threadIdx.x;
  if (i < n) out[i] = f32_to_bf16(in[i]);
}

__global__ void bias_add(const float* __restrict__ a, const float* __restrict__ b,
                         float* __restrict__ o, int n) {
  const int i = blockIdx.x * blockDim.x + threadIdx.x;
  if (i < n) o[i] = a[i] + b[i];
}

__global__ void head_gemm(const float* __restrict__ pooled,
                          const float* __restrict__ Wout,
                          const float* __restrict__ bout,
                          float* __restrict__ out) {
  const int b = threadIdx.x;
  if (b >= BATCH) return;
  float s0 = bout[0], s1 = bout[1];
  for (int k = 0; k < HIDDEN; ++k) {
    const float p = pooled[b * HIDDEN + k];
    s0 += p * Wout[k];
    s1 += p * Wout[HIDDEN + k];
  }
  out[b * 2 + 0] = s0;
  out[b * 2 + 1] = s1;
}

// ---------------------------------------------------------------------------
extern "C" void kernel_launch(void* const* d_in, const int* in_sizes, int n_in,
                              void* d_out, int out_size, void* d_ws, size_t ws_size,
                              hipStream_t stream) {
  (void)in_sizes; (void)n_in; (void)out_size; (void)ws_size;
  const int*   x    = (const int*)  d_in[0];
  const float* emb  = (const float*)d_in[1];
  const float* Wih0 = (const float*)d_in[2];
  const float* Whh0 = (const float*)d_in[3];
  const float* bih0 = (const float*)d_in[4];
  const float* bhh0 = (const float*)d_in[5];
  const float* Wih1 = (const float*)d_in[6];
  const float* Whh1 = (const float*)d_in[7];
  const float* bih1 = (const float*)d_in[8];
  const float* bhh1 = (const float*)d_in[9];
  const float* Wout = (const float*)d_in[10];
  const float* bout = (const float*)d_in[11];
  float* out = (float*)d_out;

  char* ws = (char*)d_ws;
  size_t off = 0;
  auto take = [&](size_t bytes) {
    char* p = ws + off;
    off += (bytes + 255) & ~(size_t)255;
    return p;
  };
  unsigned short* xemb  = (unsigned short*)take((size_t)BS * EMBED * 2);
  unsigned short* wih0b = (unsigned short*)take((size_t)HIDDEN * EMBED * 2);
  unsigned short* whh0b = (unsigned short*)take((size_t)HIDDEN * HIDDEN * 2);
  unsigned short* wih1b = (unsigned short*)take((size_t)HIDDEN * HIDDEN * 2);
  unsigned short* whh1b = (unsigned short*)take((size_t)HIDDEN * HIDDEN * 2);
  float*          bias0 = (float*)take((size_t)HIDDEN * 4);
  float*          bias1 = (float*)take((size_t)HIDDEN * 4);
  float*          xp    = (float*)take((size_t)BS * HIDDEN * 4);   // [S][B][H]
  unsigned short* y0    = (unsigned short*)take((size_t)BS * HIDDEN * 2);
  float*          pool  = (float*)take((size_t)BATCH * HIDDEN * 4);

  // Prep: embedding gather -> bf16, weights -> bf16, fused biases.
  embed_gather_bf16<<<(BS * EMBED / 4) / 256, 256, 0, stream>>>(x, emb, xemb);
  cvt_f32_bf16<<<(HIDDEN * EMBED  + 255) / 256, 256, 0, stream>>>(Wih0, wih0b, HIDDEN * EMBED);
  cvt_f32_bf16<<<(HIDDEN * HIDDEN + 255) / 256, 256, 0, stream>>>(Whh0, whh0b, HIDDEN * HIDDEN);
  cvt_f32_bf16<<<(HIDDEN * HIDDEN + 255) / 256, 256, 0, stream>>>(Wih1, wih1b, HIDDEN * HIDDEN);
  cvt_f32_bf16<<<(HIDDEN * HIDDEN + 255) / 256, 256, 0, stream>>>(Whh1, whh1b, HIDDEN * HIDDEN);
  bias_add<<<(HIDDEN + 255) / 256, 256, 0, stream>>>(bih0, bhh0, bias0, HIDDEN);
  bias_add<<<(HIDDEN + 255) / 256, 256, 0, stream>>>(bih1, bhh1, bias1, HIDDEN);

  const size_t rec_lds = (size_t)HIDDEN * 256 * 2      // W_hh low-K half
                       + (size_t)2 * 16 * HIDDEN * 2;  // double-buffered h
  const dim3 ggrid(BS / 256, HIDDEN / 64);

  // Layer 0: projection (time-major out), then scan storing bf16 activations.
  gemm_bf16_bias<<<ggrid, 256, 0, stream>>>(xemb, wih0b, bias0, xp, EMBED, 1);
  rnn_recurrence<true, false><<<BATCH / 16, 512, rec_lds, stream>>>(
      xp, whh0b, y0, nullptr);

  // Layer 1: projection (reuse xp), scan fused with time-mean pooling.
  gemm_bf16_bias<<<ggrid, 256, 0, stream>>>(y0, wih1b, bias1, xp, HIDDEN, 1);
  rnn_recurrence<false, true><<<BATCH / 16, 512, rec_lds, stream>>>(
      xp, whh1b, nullptr, pool);

  // Head: [64,512] @ [2,512]^T + b.
  head_gemm<<<1, 64, 0, stream>>>(pool, Wout, bout, out);
}
